// CPMAntLayer_46179488366893
// MI455X (gfx1250) — matrix operand order repaired and checked by reading, not compiled, not run on hardware
//
#include <hip/hip_runtime.h>

// ---------------------------------------------------------------------------
// CPM-Ant transformer layer for MI455X (gfx1250, wave32, WMMA bf16,
// async global->LDS staging).  B=2 S=1024 D=2048 H=16 DH=128 FF=8192
// ---------------------------------------------------------------------------

typedef unsigned short u16;
typedef __attribute__((ext_vector_type(4)))  unsigned int u32x4;
typedef __attribute__((ext_vector_type(4)))  float        f32x4;
typedef __attribute__((ext_vector_type(8)))  float        f32x8;
typedef __attribute__((ext_vector_type(16))) __bf16       bf16x16;

union Frag { u32x4 q[2]; bf16x16 v; };

#define CB   2
#define CS   1024
#define CD   2048
#define CH   16
#define CDH  128
#define CFF  8192
#define CM   (CB*CS)

__device__ __forceinline__ u16 f2bf(float f) {
    union { float f; unsigned int u; } v; v.f = f;
    unsigned int u = v.u;
    u += 0x7FFFu + ((u >> 16) & 1u);   // round-to-nearest-even
    return (u16)(u >> 16);
}

// LDS byte address of a __shared__ object: flat->LDS truncates to addr[31:0]
__device__ __forceinline__ unsigned lds_addr(const void* p) {
    return (unsigned)(unsigned long long)p;
}

// CDNA5 async copy: 16B global -> LDS, tracked by ASYNCcnt (no VGPR round trip)
__device__ __forceinline__ void async_b128(unsigned lds, const u16* g) {
    asm volatile("global_load_async_to_lds_b128 %0, %1, off"
                 :: "v"(lds), "v"((unsigned long long)g) : "memory");
}
#define WAIT_ASYNC(n) asm volatile("s_wait_asynccnt " #n ::: "memory")

// ---------------------------------------------------------------------------
// fp32 -> bf16 bulk convert (float4 in, packed 4x bf16 out)
// ---------------------------------------------------------------------------
__global__ __launch_bounds__(256) void cvt_bf16_kernel(
    const float* __restrict__ in, u16* __restrict__ out, int n4)
{
    int i = blockIdx.x * 256 + threadIdx.x;
    if (i < n4) {
        f32x4 v = ((const f32x4*)in)[i];
        unsigned long long pk =
              (unsigned long long)f2bf(v.x)
            | ((unsigned long long)f2bf(v.y) << 16)
            | ((unsigned long long)f2bf(v.z) << 32)
            | ((unsigned long long)f2bf(v.w) << 48);
        ((unsigned long long*)out)[i] = pk;
    }
}

// ---------------------------------------------------------------------------
// Tiled WMMA GEMM:  C[m,n] = alpha * sum_k A[m,k] * W[n,k]   (NT, all bf16)
// BM=BN=128, BK=32; 8 waves in 2(m) x 4(n); double-buffered LDS fed by
// global_load_async_to_lds_b128, software-pipelined with a peeled epilogue
// iteration so the steady-state loop is branch-free.
// ---------------------------------------------------------------------------
enum { MODE_QK = 0, MODE_VT = 1, MODE_O = 2, MODE_FFN1 = 3, MODE_FFN2 = 4 };

struct GemmCtx {
    const u16* ga[2]; const u16* gb[2];
    unsigned la[2][2], lb[2][2];
};

__device__ __forceinline__ void gemm_issue(GemmCtx& c, int buf) {
    #pragma unroll
    for (int i = 0; i < 2; ++i) {
        async_b128(c.la[i][buf], c.ga[i]);  c.ga[i] += 32;
        async_b128(c.lb[i][buf], c.gb[i]);  c.gb[i] += 32;
    }
}

__global__ __launch_bounds__(256) void gemm_bf16_kernel(
    const u16*  __restrict__ A, const u16* __restrict__ W,
    int M, int N, int K, float alpha, int mode,
    void* __restrict__ out0, void* __restrict__ out1,
    const float* __restrict__ bias, const float* __restrict__ resid)
{
    __shared__ u16 sA[2][128][48];   // 32 + 16 pad: 96B stride, 16B aligned
    __shared__ u16 sB[2][128][48];

    const int tid  = threadIdx.x;
    const int w    = tid >> 5;
    const int lane = tid & 31;
    const int half = lane >> 4;
    const int l15  = lane & 15;
    const int wm   = w >> 2;
    const int wn   = w & 3;
    const int bm   = blockIdx.y * 128;
    const int bn   = blockIdx.x * 128;

    // per-thread staging slots: tile = 128 rows x 32 cols bf16 = 512 x 16B
    GemmCtx c;
    #pragma unroll
    for (int i = 0; i < 2; ++i) {
        int ch = tid + i * 256;
        int r = ch >> 2, cc = (ch & 3) * 8;
        c.ga[i] = A + (size_t)(bm + r) * K + cc;
        c.gb[i] = W + (size_t)(bn + r) * K + cc;
        c.la[i][0] = lds_addr(&sA[0][r][cc]);
        c.la[i][1] = lds_addr(&sA[1][r][cc]);
        c.lb[i][0] = lds_addr(&sB[0][r][cc]);
        c.lb[i][1] = lds_addr(&sB[1][r][cc]);
    }

    f32x8 acc[4][2] = {};

    auto compute = [&](int buf) {
        Frag fa[4], fb[2];
        #pragma unroll
        for (int mt = 0; mt < 4; ++mt) {
            int r = wm * 64 + mt * 16 + l15;
            fa[mt].q[0] = *(const u32x4*)&sA[buf][r][8 * half];
            fa[mt].q[1] = *(const u32x4*)&sA[buf][r][8 * half + 16];
        }
        #pragma unroll
        for (int nt = 0; nt < 2; ++nt) {
            int r = wn * 32 + nt * 16 + l15;
            fb[nt].q[0] = *(const u32x4*)&sB[buf][r][8 * half];
            fb[nt].q[1] = *(const u32x4*)&sB[buf][r][8 * half + 16];
        }
        #pragma unroll
        for (int mt = 0; mt < 4; ++mt)
            #pragma unroll
            for (int nt = 0; nt < 2; ++nt)
                acc[mt][nt] = __builtin_amdgcn_wmma_f32_16x16x32_bf16(
                    false, fa[mt].v, false, fb[nt].v,
                    (short)0, acc[mt][nt], false, false);
    };

    const int nk = K >> 5;

    // prologue: stage tile 0 into buffer 0
    gemm_issue(c, 0);

    // steady state: branch-free — issue tile it+1, wait on tile it, compute
    for (int it = 0; it < nk - 1; ++it) {
        const int buf = it & 1;
        gemm_issue(c, buf ^ 1);
        WAIT_ASYNC(4);          // first group (tile `it`) complete (in-order)
        __syncthreads();
        compute(buf);
        __syncthreads();        // all waves done with `buf` before refill
    }
    // peeled last tile
    WAIT_ASYNC(0);
    __syncthreads();
    compute((nk - 1) & 1);

    // ---- fused epilogue
    #pragma unroll
    for (int mt = 0; mt < 4; ++mt) {
        #pragma unroll
        for (int nt = 0; nt < 2; ++nt) {
            #pragma unroll
            for (int j = 0; j < 8; ++j) {
                int gm = bm + wm * 64 + mt * 16 + 8 * half + j;
                int gn = bn + wn * 32 + nt * 16 + l15;
                float val = acc[mt][nt][j] * alpha;
                if (mode == MODE_QK || mode == MODE_VT) {
                    int b = gm >> 10, s = gm & 1023;
                    int h = gn >> 7,  d = gn & 127;
                    size_t idx = (mode == MODE_QK)
                        ? ((((size_t)b * CH + h) * CS + s) * CDH + d)
                        : ((((size_t)b * CH + h) * CDH + d) * CS + s);  // V^T
                    ((u16*)out0)[idx] = f2bf(val);
                } else if (mode == MODE_O) {
                    size_t idx = (size_t)gm * CD + gn;
                    ((float*)out0)[idx] = val;
                    ((u16*)out1)[idx]  = f2bf(val);
                } else if (mode == MODE_FFN1) {
                    float x = val + bias[gn];
                    float g = 0.5f * x * (1.0f + erff(x * 0.70710678118654752f));
                    ((u16*)out0)[(size_t)gm * N + gn] = f2bf(g);
                } else {  // MODE_FFN2: + bias + residual, fp32 out
                    size_t idx = (size_t)gm * CD + gn;
                    ((float*)out0)[idx] = val + bias[gn] + resid[idx];
                }
            }
        }
    }
}

// ---------------------------------------------------------------------------
// Fused flash attention.  grid = (S/128, H, B), 256 threads (8 waves).
// K / V^T tiles async-staged to LDS once per block (shared by all waves);
// Q A-fragments live in VGPRs; P round-trips through LDS (C->A layout).
// ---------------------------------------------------------------------------
__global__ __launch_bounds__(256) void attn_kernel(
    const u16* __restrict__ Qm, const u16* __restrict__ Km,
    const u16* __restrict__ Vt, const float* __restrict__ pb,
    const unsigned char* __restrict__ mask, u16* __restrict__ ctx)
{
    __shared__ u16 sK[64][136];    // 64 keys x 128 dh, pad 8 (272B stride)
    __shared__ u16 sV[128][72];    // 128 dh x 64 keys, pad 8 (144B stride)
    __shared__ u16 sP[128][80];    // 128 q x 64 keys, pad 16 (160B stride)

    const int tid  = threadIdx.x;
    const int w    = tid >> 5;
    const int lane = tid & 31;
    const int half = lane >> 4;
    const int l15  = lane & 15;
    const int qb = blockIdx.x, h = blockIdx.y, b = blockIdx.z;

    const u16* q0 = Qm + ((size_t)b * CH + h) * CS * CDH;
    const u16* k0 = Km + ((size_t)b * CH + h) * CS * CDH;
    const u16* v0 = Vt + ((size_t)b * CH + h) * CDH * CS;
    const int qrow0 = qb * 128 + w * 16;

    // staging slots: sK = 64x16 chunks, sV = 128x8 chunks (1024 each, 4/thread)
    const u16* gk[4]; const u16* gv[4];
    unsigned lk[4], lv[4];
    #pragma unroll
    for (int i = 0; i < 4; ++i) {
        int ch = tid + i * 256;
        int rk = ch >> 4, ck = (ch & 15) * 8;
        int rv = ch >> 3, cv = (ch & 7) * 8;
        gk[i] = k0 + (size_t)rk * CDH + ck;
        gv[i] = v0 + (size_t)rv * CS + cv;
        lk[i] = lds_addr(&sK[rk][ck]);
        lv[i] = lds_addr(&sV[rv][cv]);
    }

    // Q fragments: 16 rows x DH=128 -> 4 A-fragments resident in VGPRs
    Frag aq[4];
    #pragma unroll
    for (int kk = 0; kk < 4; ++kk) {
        const u16* p = q0 + (size_t)(qrow0 + l15) * CDH + kk * 32 + 8 * half;
        aq[kk].q[0] = *(const u32x4*)p;
        aq[kk].q[1] = *(const u32x4*)(p + 16);
    }

    f32x8 o[8] = {};
    float mr[8], lr[8];
    #pragma unroll
    for (int j = 0; j < 8; ++j) { mr[j] = -__builtin_inff(); lr[j] = 0.0f; }
    const float scale = 0.08838834764831845f;   // 1/sqrt(DH)

    for (int kt = 0; kt < CS / 64; ++kt) {
        const int kb = kt * 64;

        // ---- async-stage K and V^T tiles for this 64-key block
        #pragma unroll
        for (int i = 0; i < 4; ++i) {
            async_b128(lk[i], gk[i]);  gk[i] += (size_t)64 * CDH;  // next 64 keys
            async_b128(lv[i], gv[i]);  gv[i] += 64;
        }
        WAIT_ASYNC(0);
        __syncthreads();

        // ---- S = Q K^T  (16 x 64 per wave)
        f32x8 s[4] = {};
        #pragma unroll
        for (int tn = 0; tn < 4; ++tn) {
            #pragma unroll
            for (int kk = 0; kk < 4; ++kk) {
                Frag bK;
                const u16* p = &sK[tn * 16 + l15][kk * 32 + 8 * half];
                bK.q[0] = *(const u32x4*)p;
                bK.q[1] = *(const u32x4*)(p + 16);
                s[tn] = __builtin_amdgcn_wmma_f32_16x16x32_bf16(
                    false, aq[kk].v, false, bK.v, (short)0, s[tn], false, false);
            }
        }
        // ---- scale + position bias + mask
        #pragma unroll
        for (int tn = 0; tn < 4; ++tn) {
            int kc = kb + tn * 16 + l15;
            #pragma unroll
            for (int j = 0; j < 8; ++j) {
                int qr = qrow0 + 8 * half + j;
                float val = s[tn][j] * scale
                          + pb[((size_t)h * CS + qr) * CS + kc];
                if (!mask[((size_t)b * CS + qr) * CS + kc])
                    val = -3.0e38f;
                s[tn][j] = val;
            }
        }
        // ---- online softmax (row reduce across 16-lane half-groups)
        float corr[8];
        #pragma unroll
        for (int j = 0; j < 8; ++j) {
            float m = fmaxf(fmaxf(s[0][j], s[1][j]), fmaxf(s[2][j], s[3][j]));
            m = fmaxf(m, __shfl_xor(m, 1, 32));
            m = fmaxf(m, __shfl_xor(m, 2, 32));
            m = fmaxf(m, __shfl_xor(m, 4, 32));
            m = fmaxf(m, __shfl_xor(m, 8, 32));
            float mn = fmaxf(mr[j], m);
            corr[j] = __expf(mr[j] - mn);
            mr[j] = mn;
            float rs = 0.0f;
            #pragma unroll
            for (int tn = 0; tn < 4; ++tn) {
                float p = __expf(s[tn][j] - mn);
                s[tn][j] = p; rs += p;
            }
            rs += __shfl_xor(rs, 1, 32);
            rs += __shfl_xor(rs, 2, 32);
            rs += __shfl_xor(rs, 4, 32);
            rs += __shfl_xor(rs, 8, 32);
            lr[j] = lr[j] * corr[j] + rs;
        }
        #pragma unroll
        for (int t = 0; t < 8; ++t)
            #pragma unroll
            for (int j = 0; j < 8; ++j) o[t][j] *= corr[j];

        // ---- P (C-layout) -> LDS (wave-private rows; same-wave DS ordering)
        #pragma unroll
        for (int tn = 0; tn < 4; ++tn)
            #pragma unroll
            for (int j = 0; j < 8; ++j)
                sP[w * 16 + 8 * half + j][tn * 16 + l15] = f2bf(s[tn][j]);

        // ---- O += P V
        #pragma unroll
        for (int kk = 0; kk < 2; ++kk) {
            Frag aP;
            const u16* pp = &sP[w * 16 + l15][kk * 32 + 8 * half];
            aP.q[0] = *(const u32x4*)pp;
            aP.q[1] = *(const u32x4*)(pp + 16);
            #pragma unroll
            for (int t = 0; t < 8; ++t) {
                Frag bV;
                const u16* vp = &sV[t * 16 + l15][kk * 32 + 8 * half];
                bV.q[0] = *(const u32x4*)vp;
                bV.q[1] = *(const u32x4*)(vp + 16);
                o[t] = __builtin_amdgcn_wmma_f32_16x16x32_bf16(
                    false, aP.v, false, bV.v, (short)0, o[t], false, false);
            }
        }
        __syncthreads();   // everyone done with sK/sV before next async fill
    }

    // ---- finalize: O / l, scatter to (b, s, h*DH + d) as bf16
    float rinv[8];
    #pragma unroll
    for (int j = 0; j < 8; ++j) rinv[j] = lr[j] > 0.0f ? 1.0f / lr[j] : 0.0f;
    #pragma unroll
    for (int t = 0; t < 8; ++t) {
        #pragma unroll
        for (int j = 0; j < 8; ++j) {
            int qr = qrow0 + 8 * half + j;
            int d  = t * 16 + l15;
            ctx[((size_t)b * CS + qr) * CD + h * CDH + d] = f2bf(o[t][j] * rinv[j]);
        }
    }
}

// ---------------------------------------------------------------------------
// In-place LayerNorm over D=2048 (one block per token row)
// ---------------------------------------------------------------------------
__global__ __launch_bounds__(256) void ln_kernel(
    float* __restrict__ y, const float* __restrict__ gamma,
    const float* __restrict__ beta)
{
    __shared__ float r1[8], r2[8];
    float* x = y + (size_t)blockIdx.x * CD;
    float s1 = 0.0f, s2 = 0.0f;
    for (int i = threadIdx.x; i < CD; i += 256) {
        float v = x[i]; s1 += v; s2 += v * v;
    }
    #pragma unroll
    for (int d = 16; d >= 1; d >>= 1) {
        s1 += __shfl_xor(s1, d, 32);
        s2 += __shfl_xor(s2, d, 32);
    }
    if ((threadIdx.x & 31) == 0) { r1[threadIdx.x >> 5] = s1; r2[threadIdx.x >> 5] = s2; }
    __syncthreads();
    float t1 = 0.0f, t2 = 0.0f;
    #pragma unroll
    for (int i = 0; i < 8; ++i) { t1 += r1[i]; t2 += r2[i]; }
    float mu  = t1 * (1.0f / CD);
    float var = t2 * (1.0f / CD) - mu * mu;
    float rs  = rsqrtf(var + 1e-12f);
    for (int i = threadIdx.x; i < CD; i += 256)
        x[i] = (x[i] - mu) * rs * gamma[i] + beta[i];
}

// ---------------------------------------------------------------------------
// Orchestration
// ---------------------------------------------------------------------------
extern "C" void kernel_launch(void* const* d_in, const int* in_sizes, int n_in,
                              void* d_out, int out_size, void* d_ws, size_t ws_size,
                              hipStream_t stream)
{
    const float*         hs    = (const float*)d_in[0];
    const unsigned char* mask  = (const unsigned char*)d_in[1];
    const float*         pb    = (const float*)d_in[2];
    const float*         Wq    = (const float*)d_in[3];
    const float*         Wk    = (const float*)d_in[4];
    const float*         Wv    = (const float*)d_in[5];
    const float*         Wo    = (const float*)d_in[6];
    const float*         Wi    = (const float*)d_in[7];
    const float*         bi    = (const float*)d_in[8];
    const float*         Wd    = (const float*)d_in[9];
    const float*         bd    = (const float*)d_in[10];
    const float*         gamma = (const float*)d_in[11];
    const float*         beta  = (const float*)d_in[12];
    float* out = (float*)d_out;

    char*  ws  = (char*)d_ws;
    size_t off = 0;
    auto alloc = [&](size_t bytes) -> void* {
        void* p = ws + off;
        off += (bytes + 255) & ~(size_t)255;
        return p;
    };
    u16*   xbf    = (u16*)  alloc((size_t)CM * CD  * 2);
    u16*   wqbf   = (u16*)  alloc((size_t)CD * CD  * 2);
    u16*   wkbf   = (u16*)  alloc((size_t)CD * CD  * 2);
    u16*   wvbf   = (u16*)  alloc((size_t)CD * CD  * 2);
    u16*   wobf   = (u16*)  alloc((size_t)CD * CD  * 2);
    u16*   wibf   = (u16*)  alloc((size_t)CFF * CD * 2);
    u16*   wdbf   = (u16*)  alloc((size_t)CD * CFF * 2);
    u16*   qbf    = (u16*)  alloc((size_t)CM * CD  * 2);   // (b,h,s,d)
    u16*   kbf    = (u16*)  alloc((size_t)CM * CD  * 2);   // (b,h,s,d)
    u16*   vtbf   = (u16*)  alloc((size_t)CM * CD  * 2);   // (b,h,d,s)
    u16*   ctxbf  = (u16*)  alloc((size_t)CM * CD  * 2);   // (b,s,h*dh)
    float* attnf  = (float*)alloc((size_t)CM * CD  * 4);
    u16*   attnbf = (u16*)  alloc((size_t)CM * CD  * 2);
    u16*   hbf    = (u16*)  alloc((size_t)CM * CFF * 2);

    const float inv_sqrt_d = 0.022097086912079608f;   // 1/sqrt(2048)
    const int n4d  = CD * CD  / 4;
    const int n4f  = CFF * CD / 4;

    // 1. one-time bf16 conversion of activations + all weights
    cvt_bf16_kernel<<<(CM * CD / 4) / 256, 256, 0, stream>>>(hs, xbf, CM * CD / 4);
    cvt_bf16_kernel<<<n4d / 256, 256, 0, stream>>>(Wq, wqbf, n4d);
    cvt_bf16_kernel<<<n4d / 256, 256, 0, stream>>>(Wk, wkbf, n4d);
    cvt_bf16_kernel<<<n4d / 256, 256, 0, stream>>>(Wv, wvbf, n4d);
    cvt_bf16_kernel<<<n4d / 256, 256, 0, stream>>>(Wo, wobf, n4d);
    cvt_bf16_kernel<<<n4f / 256, 256, 0, stream>>>(Wi, wibf, n4f);
    cvt_bf16_kernel<<<n4f / 256, 256, 0, stream>>>(Wd, wdbf, n4f);

    // 2-4. QKV projections (scatter epilogues; V stored transposed)
    gemm_bf16_kernel<<<dim3(CD / 128, CM / 128), 256, 0, stream>>>(
        xbf, wqbf, CM, CD, CD, inv_sqrt_d, MODE_QK, qbf, nullptr, nullptr, nullptr);
    gemm_bf16_kernel<<<dim3(CD / 128, CM / 128), 256, 0, stream>>>(
        xbf, wkbf, CM, CD, CD, inv_sqrt_d, MODE_QK, kbf, nullptr, nullptr, nullptr);
    gemm_bf16_kernel<<<dim3(CD / 128, CM / 128), 256, 0, stream>>>(
        xbf, wvbf, CM, CD, CD, inv_sqrt_d, MODE_VT, vtbf, nullptr, nullptr, nullptr);

    // 5. fused flash attention
    attn_kernel<<<dim3(CS / 128, CH, CB), 256, 0, stream>>>(
        qbf, kbf, vtbf, pb, mask, ctxbf);

    // 6. output projection (fp32 + bf16 copies of attn_out)
    gemm_bf16_kernel<<<dim3(CD / 128, CM / 128), 256, 0, stream>>>(
        ctxbf, wobf, CM, CD, CD, inv_sqrt_d, MODE_O, attnf, attnbf, nullptr, nullptr);

    // 7. FFN up + bias + exact GELU
    gemm_bf16_kernel<<<dim3(CFF / 128, CM / 128), 256, 0, stream>>>(
        attnbf, wibf, CM, CFF, CD, 1.0f, MODE_FFN1, hbf, nullptr, bi, nullptr);

    // 8. FFN down + bias + residual -> d_out (fp32)
    gemm_bf16_kernel<<<dim3(CD / 128, CM / 128), 256, 0, stream>>>(
        hbf, wdbf, CM, CD, CFF, 1.0f, MODE_FFN2, out, nullptr, bd, attnf);

    // 9. LayerNorm in place
    ln_kernel<<<CM, 256, 0, stream>>>(out, gamma, beta);
}
